// SpikeMLP_25391846654331
// MI455X (gfx1250) — matrix operand order repaired and checked by reading, not compiled, not run on hardware
//
#include <hip/hip_runtime.h>

// ---------------------------------------------------------------------------
// SpikeMLP forward on gfx1250 (MI455X):
//   y1 = x @ w1^T + b1 ; BN(H) ; LIF -> s1 (f16, exact for {0,1})
//   y2 = s1 @ w2^T + b2 ; BN(D) ; LIF -> out (f32)
// GEMMs: v_wmma_f32_16x16x32_f16, async global->LDS staging (ASYNCcnt),
// double-buffered LDS, BN sum/sumsq fused into the GEMM epilogue (atomics).
// ---------------------------------------------------------------------------

typedef _Float16 half_t;
typedef __attribute__((ext_vector_type(16))) _Float16      v16h;
typedef __attribute__((ext_vector_type(4)))  _Float16      v4h;
typedef __attribute__((ext_vector_type(8)))  float         v8f;
typedef __attribute__((ext_vector_type(8)))  unsigned int  v8u;

#define TDIM 4
#define BDIM 32
#define NDIM 256
#define DDIM 512
#define HDIM 2048
#define ROWS (TDIM * BDIM * NDIM)                 // 32768 rows for both GEMMs
#define PLANE1 ((size_t)BDIM * NDIM * HDIM)       // 16,777,216 (per time step, layer1)
#define PLANE2 ((size_t)BDIM * NDIM * DDIM)       // 4,194,304  (per time step, layer2)

#define TAU_INV 0.5f
#define VTH 1.0f
#define BN_EPS 1e-5f

// ---------------- async global -> LDS helpers ------------------------------
__device__ __forceinline__ void async_ld_b128(unsigned lds_byte_off, const half_t* g) {
  // VGLOBAL async load: vdst = per-lane LDS byte address, vaddr = 64b global addr
  asm volatile("global_load_async_to_lds_b128 %0, %1, off"
               :: "v"(lds_byte_off), "v"(g)
               : "memory");
}

__device__ __forceinline__ void wait_async0() {
#if __has_builtin(__builtin_amdgcn_s_wait_asynccnt)
  __builtin_amdgcn_s_wait_asynccnt(0);
#else
  asm volatile("s_wait_asynccnt 0" ::: "memory");
#endif
}

// ---------------- f32 -> f16 convert (vectorized x4) -----------------------
__global__ __launch_bounds__(256) void cvt_f32_f16_x4(const float* __restrict__ in,
                                                      half_t* __restrict__ out, int n4) {
  int i = blockIdx.x * 256 + threadIdx.x;
  if (i < n4) {
    float4 f = ((const float4*)in)[i];
    v4h h;
    h.x = (half_t)f.x; h.y = (half_t)f.y; h.z = (half_t)f.z; h.w = (half_t)f.w;
    ((v4h*)out)[i] = h;
  }
}

// ---------------- zero small buffer ----------------------------------------
__global__ __launch_bounds__(256) void zero_f32(float* p, int n) {
  int i = blockIdx.x * 256 + threadIdx.x;
  if (i < n) p[i] = 0.0f;
}

// ---------------- WMMA NT GEMM + fused BN statistics -----------------------
// C[M,N] = A[M,K] * B[N,K]^T + bias[N]; stat[n] += col-sum, stat[N+n] += col-sumsq.
// Block tile 128x128, K-step 32, double-buffered async LDS staging.
// 8 waves (wave32): 2 along M x 4 along N, each wave 64x32 via 4x2 wmma frags.
__global__ __launch_bounds__(256) void gemm_nt_f16_wmma(
    const half_t* __restrict__ A, const half_t* __restrict__ B,
    const float* __restrict__ bias, float* __restrict__ C,
    float* __restrict__ stat, int M, int N, int K) {
  constexpr int BK = 32;
  constexpr int PITCH = 40;  // halves; 80B rows: 16B-aligned B128 stores, conflict-free frags
  __shared__ half_t sT[2][2][128 * PITCH];  // [buf][A|B][tile] = 40 KiB

  const int tid  = threadIdx.x;
  const int lane = tid & 31;
  const int wave = tid >> 5;
  const int wm = (wave & 1) * 64;   // wave offset along M inside block tile
  const int wn = (wave >> 1) * 32;  // wave offset along N inside block tile
  const long blockM = (long)blockIdx.y * 128;
  const long blockN = (long)blockIdx.x * 128;

  // low 32 bits of a generic LDS pointer == byte offset within LDS window
  const unsigned ldsA0 = (unsigned)(unsigned long long)(void*)&sT[0][0][0];
  const unsigned ldsB0 = (unsigned)(unsigned long long)(void*)&sT[0][1][0];
  const unsigned bufStride = (unsigned)(2 * 128 * PITCH * sizeof(half_t));

  v8f acc[4][2];
#pragma unroll
  for (int mi = 0; mi < 4; ++mi)
#pragma unroll
    for (int ni = 0; ni < 2; ++ni) {
      v8f z = {0.f, 0.f, 0.f, 0.f, 0.f, 0.f, 0.f, 0.f};
      acc[mi][ni] = z;
    }

  const int lrow = tid >> 2;        // 0..63
  const int lcol = (tid & 3) * 8;   // 0,8,16,24 halves
  const int mrow = lane & 15;       // row (A) / col (B) within 16
  const int lg   = lane >> 4;       // lane group 0/1

  // issue one tile stage (async, no VGPR round trip)
  auto stage = [&](int buf, int k0) {
    unsigned bb = bufStride * (unsigned)buf;
#pragma unroll
    for (int r = 0; r < 2; ++r) {
      int row = lrow + r * 64;
      unsigned loff = (unsigned)((row * PITCH + lcol) * sizeof(half_t));
      async_ld_b128(ldsA0 + bb + loff, A + (blockM + row) * (long)K + k0 + lcol);
      async_ld_b128(ldsB0 + bb + loff, B + (blockN + row) * (long)K + k0 + lcol);
    }
  };

  stage(0, 0);
  wait_async0();
  __syncthreads();

  int buf = 0;
  for (int k0 = 0; k0 < K; k0 += BK) {
    if (k0 + BK < K) stage(buf ^ 1, k0 + BK);  // overlap next tile with compute

    const half_t* sa = &sT[buf][0][0];
    const half_t* sb = &sT[buf][1][0];

    // ---- B fragments (32x16, K = 16*lanegroup + e) ----
    v16h bfrag[2];
#pragma unroll
    for (int ni = 0; ni < 2; ++ni) {
      const half_t* p = &sb[(wn + ni * 16 + mrow) * PITCH + 16 * lg];
      v8u u;
#pragma unroll
      for (int j = 0; j < 8; ++j) u[j] = *(const unsigned int*)(p + 2 * j);
      bfrag[ni] = __builtin_bit_cast(v16h, u);
    }

    // ---- A fragments (16x32, K = 16*lanegroup + e) ----
#pragma unroll
    for (int mi = 0; mi < 4; ++mi) {
      const half_t* p = &sa[(wm + mi * 16 + mrow) * PITCH + 16 * lg];
      v8u u;
#pragma unroll
      for (int j = 0; j < 4; ++j) u[j] = *(const unsigned int*)(p + 2 * j);
#pragma unroll
      for (int j = 4; j < 8; ++j) u[j] = *(const unsigned int*)(p + 8 + 2 * (j - 4));
      v16h afrag = __builtin_bit_cast(v16h, u);
#pragma unroll
      for (int ni = 0; ni < 2; ++ni)
        acc[mi][ni] = __builtin_amdgcn_wmma_f32_16x16x32_f16(
            false, afrag, false, bfrag[ni], (short)0, acc[mi][ni], false, false);
    }

    wait_async0();     // my async writes for next buffer have landed in LDS
    __syncthreads();   // everyone done reading current buffer + staging next
    buf ^= 1;
  }

  // ---- epilogue: store C (+bias) and fuse per-column BN sum/sumsq ----
  // C/D layout: lane L -> col N=L%16, VGPR r -> row M = r + 8*(L/16)
#pragma unroll
  for (int ni = 0; ni < 2; ++ni) {
    long col = blockN + wn + ni * 16 + mrow;
    float bv = bias[col];
    float s = 0.f, q = 0.f;
#pragma unroll
    for (int mi = 0; mi < 4; ++mi) {
      long rowBase = blockM + wm + mi * 16 + 8 * lg;
      v8f c = acc[mi][ni];
#pragma unroll
      for (int r = 0; r < 8; ++r) {
        float v = c[r] + bv;
        C[(rowBase + r) * (long)N + col] = v;
        s += v;
        q += v * v;
      }
    }
    // lane L and L+16 hold the same column's other 8-row half
    s += __shfl_xor(s, 16, 32);
    q += __shfl_xor(q, 16, 32);
    if (lg == 0) {
      atomicAdd(&stat[col], s);
      atomicAdd(&stat[N + col], q);
    }
  }
}

// ---------------- fused BN(affine) + LIF over T ----------------------------
// y: [T, plane] f32 ; out: [T, plane] OutT (f16 spikes for layer1, f32 final)
template <typename OutT>
__global__ __launch_bounds__(256) void bn_lif(const float* y, const float* stat,
                                              const float* gamma, const float* beta,
                                              OutT* out, int C, size_t plane,
                                              float invCnt) {
  size_t i = (size_t)blockIdx.x * 256 + threadIdx.x;
  if (i >= plane) return;
  int c = (int)(i % (size_t)C);
  float m   = stat[c] * invCnt;
  float var = stat[C + c] * invCnt - m * m;
  float rs = rsqrtf(var + BN_EPS);
  float ga = gamma[c] * rs;
  float bb = beta[c] - m * ga;
  float v = 0.f;
#pragma unroll
  for (int t = 0; t < TDIM; ++t) {
    float val = y[(size_t)t * plane + i] * ga + bb;   // BN
    v += (val - v) * TAU_INV;                          // charge
    float sp = (v - VTH >= 0.f) ? 1.f : 0.f;           // fire
    v *= (1.f - sp);                                   // hard reset
    out[(size_t)t * plane + i] = (OutT)sp;
  }
}

// ---------------------------------------------------------------------------
extern "C" void kernel_launch(void* const* d_in, const int* in_sizes, int n_in,
                              void* d_out, int out_size, void* d_ws, size_t ws_size,
                              hipStream_t stream) {
  const float* x   = (const float*)d_in[0];
  const float* w1  = (const float*)d_in[1];
  const float* b1  = (const float*)d_in[2];
  const float* g1  = (const float*)d_in[3];
  const float* be1 = (const float*)d_in[4];
  const float* w2  = (const float*)d_in[5];
  const float* b2  = (const float*)d_in[6];
  const float* g2  = (const float*)d_in[7];
  const float* be2 = (const float*)d_in[8];
  float* out = (float*)d_out;

  char* ws = (char*)d_ws;
  half_t* xh  = (half_t*)ws; ws += (size_t)ROWS * DDIM * sizeof(half_t);   // 32 MiB
  half_t* w1h = (half_t*)ws; ws += (size_t)HDIM * DDIM * sizeof(half_t);   //  2 MiB
  half_t* w2h = (half_t*)ws; ws += (size_t)DDIM * HDIM * sizeof(half_t);   //  2 MiB
  float*  y1  = (float*)ws;  ws += (size_t)ROWS * HDIM * sizeof(float);    // 256 MiB
  half_t* s1h = (half_t*)ws; ws += (size_t)ROWS * HDIM * sizeof(half_t);   // 128 MiB
  float*  st  = (float*)ws;  // 2*HDIM + 2*DDIM floats
  float*  st1 = st;
  float*  st2 = st + 2 * HDIM;

  // f32 -> f16 converts
  cvt_f32_f16_x4<<<(ROWS * DDIM / 4) / 256, 256, 0, stream>>>(x, xh, ROWS * DDIM / 4);
  cvt_f32_f16_x4<<<(HDIM * DDIM / 4) / 256, 256, 0, stream>>>(w1, w1h, HDIM * DDIM / 4);
  cvt_f32_f16_x4<<<(DDIM * HDIM / 4) / 256, 256, 0, stream>>>(w2, w2h, DDIM * HDIM / 4);

  // zero BN stats (must happen every call; harness replays the graph)
  zero_f32<<<(2 * HDIM + 2 * DDIM + 255) / 256, 256, 0, stream>>>(st, 2 * HDIM + 2 * DDIM);

  // layer 1: GEMM + bias + stats -> y1/st1 ; BN+LIF -> f16 spikes
  gemm_nt_f16_wmma<<<dim3(HDIM / 128, ROWS / 128), 256, 0, stream>>>(
      xh, w1h, b1, y1, st1, ROWS, HDIM, DDIM);
  bn_lif<half_t><<<(unsigned)(PLANE1 / 256), 256, 0, stream>>>(
      y1, st1, g1, be1, s1h, HDIM, PLANE1, 1.0f / (float)ROWS);

  // layer 2: GEMM + bias + stats -> d_out/st2 ; BN+LIF in place
  gemm_nt_f16_wmma<<<dim3(DDIM / 128, ROWS / 128), 256, 0, stream>>>(
      s1h, w2h, b2, out, st2, ROWS, DDIM, HDIM);
  bn_lif<float><<<(unsigned)(PLANE2 / 256), 256, 0, stream>>>(
      out, st2, g2, be2, out, DDIM, PLANE2, 1.0f / (float)ROWS);
}
